// VanillaRNN_77756087927458
// MI455X (gfx1250) — matrix operand order repaired and checked by reading, not compiled, not run on hardware
//
#include <hip/hip_runtime.h>
#include <hip/hip_bf16.h>

#define VOCAB 32000
#define HID   1024
#define SEQ   512
#define BATCH 8
#define NROW  (BATCH*SEQ)   // 4096
#define RNN_NWG 64

typedef __attribute__((ext_vector_type(2)))  float  v2f;
typedef __attribute__((ext_vector_type(8)))  float  v8f;
typedef __attribute__((ext_vector_type(16))) __bf16 v16bf;

static __device__ __forceinline__ unsigned short f32_to_bf16(float f) {
  union { float f; unsigned int u; } x; x.f = f;
  unsigned int r = x.u + 0x7fffu + ((x.u >> 16) & 1u);   // round-to-nearest-even
  return (unsigned short)(r >> 16);
}

union BF16Frag { uint4 q[2]; v16bf v; };

// A fragment (16x32 bf16): lane-half g: K = {8g..8g+7} U {8g+16..8g+23}
static __device__ __forceinline__ v16bf load_frag_a_bf16(const unsigned short* rowK, int g) {
  BF16Frag f;
  f.q[0] = *(const uint4*)(rowK + 8*g);
  f.q[1] = *(const uint4*)(rowK + 8*g + 16);
  return f.v;
}
// B fragment (32x16 bf16): lane-half g: K = {8g..8g+7} U {8g+16..8g+23}
static __device__ __forceinline__ v16bf load_frag_b_bf16(const unsigned short* rowK, int g) {
  BF16Frag f;
  f.q[0] = *(const uint4*)(rowK + 8*g);
  f.q[1] = *(const uint4*)(rowK + 8*g + 16);
  return f.v;
}

// ---------------------------------------------------------------------------
// Kernel 0: per-call state init (barrier counters + h0 = 0). Deterministic
// under graph replay.
// ---------------------------------------------------------------------------
__global__ __launch_bounds__(1024)
void k_init(float* __restrict__ hA, int* __restrict__ cnt) {
  int i = threadIdx.x;
  if (i < SEQ) cnt[i] = 0;
  for (int j = i; j < BATCH*HID; j += 1024) hA[j] = 0.0f;
}

// ---------------------------------------------------------------------------
// Kernel 1a: f32 transpose  in[R][C] -> outT[C][R]   (32x32 tiles, 33-pad)
// Used for w_x and w_h so GEMM B-fragments become contiguous ds_load_b64.
// ---------------------------------------------------------------------------
__global__ __launch_bounds__(256)
void k_transpose_f32(const float* __restrict__ in, float* __restrict__ outT,
                     int R, int C) {
  __shared__ float tile[32][33];
  const int x  = threadIdx.x & 31;
  const int y0 = threadIdx.x >> 5;     // 0..7
  const int c0 = blockIdx.x * 32;
  const int r0 = blockIdx.y * 32;
#pragma unroll
  for (int it = 0; it < 4; ++it) {
    int y = y0 + it * 8;
    tile[y][x] = in[(size_t)(r0 + y) * C + c0 + x];
  }
  __syncthreads();
#pragma unroll
  for (int it = 0; it < 4; ++it) {
    int y = y0 + it * 8;
    outT[(size_t)(c0 + y) * R + r0 + x] = tile[x][y];
  }
}

// ---------------------------------------------------------------------------
// Kernel 1b: fused transpose + f32->bf16 convert: w_out[1024][32000] ->
// w_outT16[32000][1024].  Makes the big GEMM's B staging identical in shape
// to A staging (coalesced uint4 both sides, no LDS bank conflicts).
// ---------------------------------------------------------------------------
__global__ __launch_bounds__(256)
void k_transpose_cvt_bf16(const float* __restrict__ in,
                          unsigned short* __restrict__ outT, int R, int C) {
  __shared__ float tile[32][33];
  const int x  = threadIdx.x & 31;
  const int y0 = threadIdx.x >> 5;
  const int c0 = blockIdx.x * 32;
  const int r0 = blockIdx.y * 32;
#pragma unroll
  for (int it = 0; it < 4; ++it) {
    int y = y0 + it * 8;
    tile[y][x] = in[(size_t)(r0 + y) * C + c0 + x];
  }
  __syncthreads();
#pragma unroll
  for (int it = 0; it < 4; ++it) {
    int y = y0 + it * 8;
    outT[(size_t)(c0 + y) * R + r0 + x] = f32_to_bf16(tile[x][y]);
  }
}

// ---------------------------------------------------------------------------
// Kernel 2: xw[m,n] = emb[tokens[m],:] @ w_x + b_h   (f32 WMMA 16x16x4)
// B read from pre-transposed w_xT[n][k]; both fragments are ds_load_b64.
// ---------------------------------------------------------------------------
__global__ __launch_bounds__(256)
void k_embed_xproj(const int* __restrict__ tokens, const float* __restrict__ emb,
                   const float* __restrict__ w_xT, const float* __restrict__ b_h,
                   float* __restrict__ xw) {
  __shared__ int   sTok[128];
  __shared__ __align__(16) float sA[128][32];
  __shared__ __align__(16) float sBt[64][32];     // [n][k]
  const int tid  = threadIdx.x;
  const int lane = tid & 31, wv = tid >> 5;
  const int g = lane >> 4, nl = lane & 15;
  const int m0 = blockIdx.y * 128;
  const int n0 = blockIdx.x * 64;
  if (tid < 128) sTok[tid] = tokens[m0 + tid];
  __syncthreads();

  v8f acc[4] = {};
  for (int kc = 0; kc < HID; kc += 32) {
    // Stage gathered A rows (embedding lookup fused into the GEMM)
#pragma unroll
    for (int it = 0; it < 4; ++it) {
      int idx = tid + it * 256;             // 0..1023
      int r = idx >> 3, q = idx & 7;        // r<128, q<8 (4 floats)
      *(uint4*)&sA[r][q * 4] =
          *(const uint4*)&emb[(size_t)sTok[r] * HID + kc + q * 4];
    }
    // Stage B from w_xT (already [n][k]) - coalesced, conflict-free
#pragma unroll
    for (int it = 0; it < 2; ++it) {
      int idx = tid + it * 256;             // 0..511
      int r = idx >> 3, q = idx & 7;        // r<64 (n), q<8 (4 floats of k)
      *(uint4*)&sBt[r][q * 4] =
          *(const uint4*)&w_xT[(size_t)(n0 + r) * HID + kc + q * 4];
    }
    __syncthreads();
#pragma unroll
    for (int ks = 0; ks < 8; ++ks) {
      int kb = ks * 4 + 2 * g;
      v2f a = *(const v2f*)&sA[wv * 16 + nl][kb];
#pragma unroll
      for (int nt = 0; nt < 4; ++nt) {
        v2f b = *(const v2f*)&sBt[nt * 16 + nl][kb];
        acc[nt] = __builtin_amdgcn_wmma_f32_16x16x4_f32(
            false, a, false, b, (short)0, acc[nt], false, false);
      }
    }
    __syncthreads();
  }
#pragma unroll
  for (int nt = 0; nt < 4; ++nt) {
    int n = n0 + nt * 16 + nl;
    float bo = b_h[n];
#pragma unroll
    for (int r = 0; r < 8; ++r) {
      int m = m0 + wv * 16 + g * 8 + r;
      xw[(size_t)m * HID + n] = acc[nt][r] + bo;
    }
  }
}

// ---------------------------------------------------------------------------
// Kernel 3: sequential scan  h = tanh(xw_t + h @ w_h),  512 steps.
// 64 persistent WGs x 128 thr (4 waves). WG keeps its w_hT column slice
// ([16 cols][1024 k] f32 = 64KB) resident in LDS for the whole scan; both
// WMMA fragments are contiguous ds_load_b64.  Per-step grid barrier on a
// dedicated counter.
// ---------------------------------------------------------------------------
__global__ __launch_bounds__(128)
void k_rnn_scan(const float* __restrict__ xw, const float* __restrict__ w_hT,
                float* __restrict__ hA, float* __restrict__ hB,
                unsigned short* __restrict__ hs16, int* __restrict__ cnt) {
  extern __shared__ float smem[];
  float* sW = smem;                 // [16][1024] = 64KB   (col-major slice)
  float* sH = smem + 16 * 1024;     // [8][1024]  = 32KB
  float* sR = sH + 8 * 1024;        // [4][16][16] = 4KB
  const int tid  = threadIdx.x;
  const int lane = tid & 31, wv = tid >> 5;
  const int g = lane >> 4, nl = lane & 15;
  const int n0 = blockIdx.x * 16;

  // Preload w_hT rows n0..n0+15 (contiguous k) - coalesced, conflict-free
  for (int it = 0; it < 32; ++it) {
    int idx = tid + it * 128;       // 0..4095
    int c = idx >> 8, q = idx & 255;
    *(uint4*)&sW[c * 1024 + q * 4] =
        *(const uint4*)&w_hT[(size_t)(n0 + c) * HID + q * 4];
  }
  __syncthreads();

  for (int t = 0; t < SEQ; ++t) {
    const float* hprev = (t & 1) ? hB : hA;
    float*       hnext = (t & 1) ? hA : hB;
#pragma unroll
    for (int it = 0; it < 16; ++it) {
      int idx = tid + it * 128;     // 0..2047 (uint4 units)
      *(uint4*)&sH[idx * 4] = *(const uint4*)&hprev[idx * 4];
    }
    __syncthreads();

    v8f acc = {};
    const int kb0 = wv * 256 + 2 * g;
#pragma unroll 8
    for (int kt = 0; kt < 64; ++kt) {
      int kb = kb0 + kt * 4;
      v2f a;
      if (nl < 8) a = *(const v2f*)&sH[nl * HID + kb];   // pad M 8->16 w/ zeros
      else { a.x = 0.0f; a.y = 0.0f; }
      v2f b = *(const v2f*)&sW[nl * 1024 + kb];
      acc = __builtin_amdgcn_wmma_f32_16x16x4_f32(
          false, a, false, b, (short)0, acc, false, false);
    }
#pragma unroll
    for (int r = 0; r < 8; ++r)
      sR[(wv * 16 + g * 8 + r) * 16 + nl] = acc[r];
    __syncthreads();

    if (wv == 0 && g == 0) {        // 16 lanes finish columns n0..n0+15
#pragma unroll
      for (int m = 0; m < 8; ++m) {
        float s = sR[(0 * 16 + m) * 16 + nl] + sR[(1 * 16 + m) * 16 + nl] +
                  sR[(2 * 16 + m) * 16 + nl] + sR[(3 * 16 + m) * 16 + nl];
        size_t row = ((size_t)m * SEQ + t) * HID + n0 + nl;
        float v = tanhf(xw[row] + s);
        hnext[m * HID + n0 + nl] = v;
        hs16[row] = f32_to_bf16(v);
      }
    }
    if (tid == 0) {
      __threadfence();
      __hip_atomic_fetch_add(&cnt[t], 1, __ATOMIC_ACQ_REL, __HIP_MEMORY_SCOPE_AGENT);
      while (__hip_atomic_load(&cnt[t], __ATOMIC_ACQUIRE, __HIP_MEMORY_SCOPE_AGENT) < RNN_NWG)
        __builtin_amdgcn_s_sleep(2);
      __threadfence();
    }
    __syncthreads();
  }
}

// ---------------------------------------------------------------------------
// Kernel 4: logits = hs @ w_out + b_out  (bf16 WMMA 16x16x32, f32 accum)
// B read from pre-transposed w_outT16[n][k]: staging is the same coalesced
// uint4 pattern as A; fragments are two 16B LDS reads each.
// ---------------------------------------------------------------------------
__global__ __launch_bounds__(256)
void k_out_gemm(const unsigned short* __restrict__ A16,    // hs [4096][1024]
                const unsigned short* __restrict__ B16T,   // w_outT [32000][1024]
                const float* __restrict__ b_out,
                float* __restrict__ out) {
  __shared__ __align__(16) unsigned short sA[256][32];
  __shared__ __align__(16) unsigned short sB[128][32];     // [n][k]
  const int tid  = threadIdx.x;
  const int lane = tid & 31, wv = tid >> 5;
  const int g = lane >> 4, nl = lane & 15;
  const int wm = wv & 3, wn = wv >> 2;
  const int n0 = blockIdx.x * 128;
  const int m0 = blockIdx.y * 256;

  v8f zero = {};
  v8f acc[4][4];
#pragma unroll
  for (int i = 0; i < 4; ++i)
#pragma unroll
    for (int j = 0; j < 4; ++j) acc[i][j] = zero;

  for (int kc = 0; kc < HID; kc += 32) {
#pragma unroll
    for (int it = 0; it < 4; ++it) {
      int idx = tid + it * 256;         // 0..1023
      int r = idx >> 2, q = idx & 3;    // r<256, q<4 (8 bf16)
      *(uint4*)&sA[r][q * 8] =
          *(const uint4*)&A16[(size_t)(m0 + r) * HID + kc + q * 8];
    }
#pragma unroll
    for (int it = 0; it < 2; ++it) {
      int idx = tid + it * 256;         // 0..511
      int r = idx >> 2, q = idx & 3;    // r<128 (n), q<4 (8 bf16)
      *(uint4*)&sB[r][q * 8] =
          *(const uint4*)&B16T[(size_t)(n0 + r) * HID + kc + q * 8];
    }
    __syncthreads();

    v16bf af[4];
#pragma unroll
    for (int mt = 0; mt < 4; ++mt)
      af[mt] = load_frag_a_bf16(&sA[wm * 64 + mt * 16 + nl][0], g);
#pragma unroll
    for (int nt = 0; nt < 4; ++nt) {
      v16bf bf = load_frag_b_bf16(&sB[wn * 64 + nt * 16 + nl][0], g);
#pragma unroll
      for (int mt = 0; mt < 4; ++mt)
        acc[mt][nt] = __builtin_amdgcn_wmma_f32_16x16x32_bf16(
            false, af[mt], false, bf, (short)0, acc[mt][nt], false, false);
    }
    __syncthreads();
  }

#pragma unroll
  for (int nt = 0; nt < 4; ++nt) {
    int n = n0 + wn * 64 + nt * 16 + nl;
    float bo = b_out[n];
#pragma unroll
    for (int mt = 0; mt < 4; ++mt) {
#pragma unroll
      for (int r = 0; r < 8; ++r) {
        int m = m0 + wm * 64 + mt * 16 + g * 8 + r;
        out[(size_t)m * VOCAB + n] = acc[mt][nt][r] + bo;
      }
    }
  }
}

// ---------------------------------------------------------------------------
// Workspace layout (bytes):                       total ~94.6 MB
//   xw       f32 [4096][1024]      @ 0          (16,777,216)
//   hA       f32 [8][1024]         @ 16777216   (32,768)
//   hB       f32 [8][1024]         @ 16809984   (32,768)
//   cnt      int [512]             @ 16842752   (4,096 padded)
//   hs16     bf16 [4096][1024]     @ 16846848   (8,388,608)
//   woutT16  bf16 [32000][1024]    @ 25235456   (65,536,000)
//   w_xT     f32 [1024][1024]      @ 90771456   (4,194,304)
//   w_hT     f32 [1024][1024]      @ 94965760   (4,194,304)
// ---------------------------------------------------------------------------
extern "C" void kernel_launch(void* const* d_in, const int* in_sizes, int n_in,
                              void* d_out, int out_size, void* d_ws, size_t ws_size,
                              hipStream_t stream) {
  (void)in_sizes; (void)n_in; (void)out_size; (void)ws_size;
  const int*   tokens = (const int*)d_in[0];
  const float* emb    = (const float*)d_in[1];
  const float* w_x    = (const float*)d_in[2];
  const float* w_h    = (const float*)d_in[3];
  const float* b_h    = (const float*)d_in[4];
  const float* w_out  = (const float*)d_in[5];
  const float* b_out  = (const float*)d_in[6];
  float* out = (float*)d_out;

  char* ws = (char*)d_ws;
  float*          xw      = (float*)(ws + 0);
  float*          hA      = (float*)(ws + 16777216);
  float*          hB      = (float*)(ws + 16809984);
  int*            cnt     = (int*)  (ws + 16842752);
  unsigned short* hs16    = (unsigned short*)(ws + 16846848);
  unsigned short* woutT16 = (unsigned short*)(ws + 25235456);
  float*          w_xT    = (float*)(ws + 90771456);
  float*          w_hT    = (float*)(ws + 94965760);

  k_init<<<dim3(1), dim3(1024), 0, stream>>>(hA, cnt);
  // One-time weight reshapes (139MB touched once, ~6us at HBM peak)
  k_transpose_f32<<<dim3(HID / 32, HID / 32), dim3(256), 0, stream>>>(
      w_x, w_xT, HID, HID);
  k_transpose_f32<<<dim3(HID / 32, HID / 32), dim3(256), 0, stream>>>(
      w_h, w_hT, HID, HID);
  k_transpose_cvt_bf16<<<dim3(VOCAB / 32, HID / 32), dim3(256), 0, stream>>>(
      w_out, woutT16, HID, VOCAB);
  // Pipeline
  k_embed_xproj<<<dim3(HID / 64, NROW / 128), dim3(256), 0, stream>>>(
      tokens, emb, w_xT, b_h, xw);
  k_rnn_scan<<<dim3(RNN_NWG), dim3(128), 102400 /* 64K W + 32K H + 4K R */, stream>>>(
      xw, w_hT, hA, hB, hs16, cnt);
  k_out_gemm<<<dim3(VOCAB / 128, NROW / 256), dim3(256), 0, stream>>>(
      hs16, woutT16, b_out, out);
}
